// pLoss_all_fidelity_85907935855434
// MI455X (gfx1250) — matrix operand order repaired and checked by reading, not compiled, not run on hardware
//
#include <hip/hip_runtime.h>

#define NSTATES 16384
#define NBATCH  4096
#define NC      64

typedef _Float16 v16h __attribute__((ext_vector_type(16)));
typedef float    v8f  __attribute__((ext_vector_type(8)));
typedef float    f4v  __attribute__((ext_vector_type(4)));

union V16U { v16h h; f4v v[2]; };

// 16 contiguous halves (32B) -> v16h
__device__ __forceinline__ v16h load16_contig(const _Float16* p) {
    V16U u;
    u.v[0] = *(const f4v*)(p);
    u.v[1] = *(const f4v*)(p + 8);
    return u.h;
}
// 8 halves at p and 8 halves at p+16 (A-operand split layout)
__device__ __forceinline__ v16h load16_split(const _Float16* p) {
    V16U u;
    u.v[0] = *(const f4v*)(p);
    u.v[1] = *(const f4v*)(p + 16);
    return u.h;
}

// Prologue: S(int32) -> f16 row-major and f16 transposed; f(fp32) -> f16.
__global__ void k_convert(const float* __restrict__ f, const int* __restrict__ S,
                          _Float16* __restrict__ f16f, _Float16* __restrict__ s16,
                          _Float16* __restrict__ st16) {
    int idx = blockIdx.x * 256 + threadIdx.x;
    if (idx < NSTATES * NC) {
        _Float16 h = (_Float16)(float)S[idx];
        s16[idx] = h;
        int s = idx >> 6, c = idx & 63;
        st16[(size_t)c * NSTATES + s] = h;
    }
    if (idx < NBATCH * NC) f16f[idx] = (_Float16)f[idx];
}

// Main fused kernel: one workgroup = 16 batch columns, 8 waves split the 16384 states.
__global__ __launch_bounds__(256) void k_main(const _Float16* __restrict__ f16f,
                                              const _Float16* __restrict__ s16,
                                              const _Float16* __restrict__ st16,
                                              const int* __restrict__ y,
                                              float* __restrict__ out,
                                              float* __restrict__ partials) {
    __shared__ float wred[8][32];
    __shared__ float mfin[16];
    __shared__ float zfin[16];
    __shared__ float snum8[8][1024];   // per-wave [16 batch x 64 class] partial numerators
    __shared__ float sl[256];

    const int tid  = threadIdx.x;
    const int lane = tid & 31;
    const int hl   = lane >> 4;   // lane half
    const int ln   = lane & 15;
    // wave index is wave-uniform: pin it to an SGPR so the state loops are scalar loops
    const int wave = __builtin_amdgcn_readfirstlane(tid >> 5);
    const int batchBase = blockIdx.x * 16;

    // Per-iteration strides (in halves): 8 waves x 32 states round-robin
    const size_t A_STRIDE = (size_t)8 * 32 * NC;   // 16384 halves = 32 KiB
    const size_t B_STRIDE = (size_t)8 * 32;        // 256 halves   = 512 B
    const int NITER = NSTATES / (8 * 32);          // 64 uniform iterations per wave

    // B operand of GEMM1: f-tile (K=classes, N=16 batch), two K=32 tiles.
    const _Float16* fb = f16f + (size_t)(batchBase + ln) * NC + hl * 8;
    v16h Bf0 = load16_split(fb);
    v16h Bf1 = load16_split(fb + 32);

    // ---- Pass 1: per-batch max of potential ----
    float lmax = -3.0e38f;
    {
        const _Float16* ap = s16 + ((size_t)(wave * 32) + ln) * NC + hl * 8;
        for (int it = 0; it < NITER; ++it, ap += A_STRIDE) {
            __builtin_prefetch(ap + A_STRIDE, 0, 1);
            v16h A10 = load16_split(ap);           // states 0-15, classes 0-31
            v16h A11 = load16_split(ap + 32);      // states 0-15, classes 32-63
            v16h A20 = load16_split(ap + 1024);    // states 16-31 (+16*64 halves)
            v16h A21 = load16_split(ap + 1056);
            v8f c1 = {}, c2 = {};
            c1 = __builtin_amdgcn_wmma_f32_16x16x32_f16(false, A10, false, Bf0, (short)0, c1, false, false);
            c1 = __builtin_amdgcn_wmma_f32_16x16x32_f16(false, A11, false, Bf1, (short)0, c1, false, false);
            c2 = __builtin_amdgcn_wmma_f32_16x16x32_f16(false, A20, false, Bf0, (short)0, c2, false, false);
            c2 = __builtin_amdgcn_wmma_f32_16x16x32_f16(false, A21, false, Bf1, (short)0, c2, false, false);
#pragma unroll
            for (int r = 0; r < 8; ++r) lmax = fmaxf(lmax, fmaxf(c1[r], c2[r]));
        }
    }
    wred[wave][lane] = lmax;
    __syncthreads();
    if (tid < 16) {
        float m = -3.0e38f;
#pragma unroll
        for (int w = 0; w < 8; ++w) m = fmaxf(m, fmaxf(wred[w][tid], wred[w][tid + 16]));
        mfin[tid] = m;
    }
    __syncthreads();
    const float m = mfin[ln];   // C-tile column of this lane is batch ln

    // ---- Pass 2: z and num = J^T * S via second WMMA chain ----
    float zp = 0.f;
    v8f num0 = {}, num1 = {}, num2 = {}, num3 = {};
    {
        const _Float16* ap = s16 + ((size_t)(wave * 32) + ln) * NC + hl * 8;
        const _Float16* bp = st16 + (size_t)ln * NSTATES + (size_t)(wave * 32) + hl * 8;
        for (int it = 0; it < NITER; ++it, ap += A_STRIDE, bp += B_STRIDE) {
            __builtin_prefetch(ap + A_STRIDE, 0, 1);
            v16h A10 = load16_split(ap);
            v16h A11 = load16_split(ap + 32);
            v16h A20 = load16_split(ap + 1024);
            v16h A21 = load16_split(ap + 1056);
            v8f c1 = {}, c2 = {};
            c1 = __builtin_amdgcn_wmma_f32_16x16x32_f16(false, A10, false, Bf0, (short)0, c1, false, false);
            c1 = __builtin_amdgcn_wmma_f32_16x16x32_f16(false, A11, false, Bf1, (short)0, c1, false, false);
            c2 = __builtin_amdgcn_wmma_f32_16x16x32_f16(false, A20, false, Bf0, (short)0, c2, false, false);
            c2 = __builtin_amdgcn_wmma_f32_16x16x32_f16(false, A21, false, Bf1, (short)0, c2, false, false);

            // J = exp(pot - m); in-lane transpose C-layout -> A-layout (no shuffles)
            v16h Aj;
#pragma unroll
            for (int r = 0; r < 8; ++r) {
                float j1 = __expf(c1[r] - m);
                float j2 = __expf(c2[r] - m);
                zp += j1 + j2;
                Aj[r]     = (_Float16)j1;   // K(local) 0-7 (lanes<16) / 8-15 (lanes>=16)
                Aj[8 + r] = (_Float16)j2;   // K(local) 16-23 / 24-31
            }

            // B operand of GEMM2: S^T tiles (K=32 states, N=16 classes), 4 class tiles
            v16h B20 = load16_split(bp + 0 * 16 * (size_t)NSTATES);
            v16h B21 = load16_split(bp + 1 * 16 * (size_t)NSTATES);
            v16h B22 = load16_split(bp + 2 * 16 * (size_t)NSTATES);
            v16h B23 = load16_split(bp + 3 * 16 * (size_t)NSTATES);
            num0 = __builtin_amdgcn_wmma_f32_16x16x32_f16(false, Aj, false, B20, (short)0, num0, false, false);
            num1 = __builtin_amdgcn_wmma_f32_16x16x32_f16(false, Aj, false, B21, (short)0, num1, false, false);
            num2 = __builtin_amdgcn_wmma_f32_16x16x32_f16(false, Aj, false, B22, (short)0, num2, false, false);
            num3 = __builtin_amdgcn_wmma_f32_16x16x32_f16(false, Aj, false, B23, (short)0, num3, false, false);
        }
    }
    wred[wave][lane] = zp;
    __syncthreads();
    if (tid < 16) {
        float z = 0.f;
#pragma unroll
        for (int w = 0; w < 8; ++w) z += wred[w][tid] + wred[w][tid + 16];
        zfin[tid] = z;
    }
    // dump per-wave num tiles (deterministic; no atomics)
#pragma unroll
    for (int r = 0; r < 8; ++r) {
        const int row = (r + hl * 8) * 64 + ln;  // (batch M)*64 + class N
        snum8[wave][row +  0] = num0[r];
        snum8[wave][row + 16] = num1[r];
        snum8[wave][row + 32] = num2[r];
        snum8[wave][row + 48] = num3[r];
    }
    __syncthreads();

    // ---- Pass 3: pMargin + focal BCE ----
    float lpart = 0.f;
    for (int i = tid; i < 1024; i += 256) {
        float nsum = 0.f;
#pragma unroll
        for (int w = 0; w < 8; ++w) nsum += snum8[w][i];
        const int b = i >> 6, c = i & 63;
        const float p = nsum / zfin[b];
        out[1 + (size_t)(batchBase + b) * NC + c] = p;
        const int yv = y[(size_t)(batchBase + b) * NC + c];
        const float yf = (float)yv;
        const float log_p   = fmaxf(__logf(p), -100.f);
        const float log_1mp = fmaxf(log1pf(-p), -100.f);
        const float bce = -(yf * log_p + (1.f - yf) * log_1mp);
        const float p_t = yv ? p : 1.f - p;
        const float fw  = __fsqrt_rn(fmaxf(1.f - p_t, 0.f));   // gamma = 0.5
        const float alpha_t = yv ? 10.f : -9.f;                // alpha=10, 1-alpha=-9
        lpart += fw * alpha_t * bce;
    }
    sl[tid] = lpart;
    __syncthreads();
    if (tid == 0) {
        float t = 0.f;
        for (int i = 0; i < 256; ++i) t += sl[i];
        partials[blockIdx.x] = t;
    }
}

// Deterministic final reduction of the 256 per-workgroup loss partials.
__global__ void k_final(const float* __restrict__ partials, float* __restrict__ out) {
    if (threadIdx.x == 0) {
        float t = 0.f;
        for (int i = 0; i < NBATCH / 16; ++i) t += partials[i];
        out[0] = t * (1.0f / ((float)NBATCH * (float)NC));
    }
}

extern "C" void kernel_launch(void* const* d_in, const int* in_sizes, int n_in,
                              void* d_out, int out_size, void* d_ws, size_t ws_size,
                              hipStream_t stream) {
    const float* f = (const float*)d_in[0];
    const int*   S = (const int*)d_in[1];
    const int*   yy = (const int*)d_in[2];
    float* out = (float*)d_out;
    char* ws = (char*)d_ws;

    _Float16* f16f = (_Float16*)(ws);                                  // 512 KB
    _Float16* s16  = (_Float16*)(ws + 512 * 1024);                     // 2 MB
    _Float16* st16 = (_Float16*)(ws + 512 * 1024 + 2 * 1024 * 1024);   // 2 MB
    float* partials = (float*)(ws + 512 * 1024 + 4 * 1024 * 1024);     // 1 KB

    k_convert<<<(NSTATES * NC + 255) / 256, 256, 0, stream>>>(f, S, f16f, s16, st16);
    k_main<<<NBATCH / 16, 256, 0, stream>>>(f16f, s16, st16, yy, out, partials);
    k_final<<<1, 64, 0, stream>>>(partials, out);
}
